// MPNNBlock_65335042506829
// MI455X (gfx1250) — matrix operand order repaired and checked by reading, not compiled, not run on hardware
//
#include <hip/hip_runtime.h>

typedef __bf16 bf16_t;
typedef __attribute__((ext_vector_type(16))) __bf16 v16bf;
typedef __attribute__((ext_vector_type(8)))  __bf16 v8bf;
typedef __attribute__((ext_vector_type(8)))  float  v8f;

#define NNODES 16384
#define KNBR   32
#define LDIM   128
#define PDIM   64
#define CINF   320
#define H1DIM  512
#define H2DIM  512
#define HPDIM  128

union V16U { v16bf v; v8bf h[2]; };

__device__ __forceinline__ v8f wmma_bf16(v16bf a, v16bf b, v8f c) {
  return __builtin_amdgcn_wmma_f32_16x16x32_bf16(false, a, false, b, (short)0, c, false, false);
}

// A fragment (16x32 bf16) from LDS row-major buffer.
// lane: m = lane%16, half = lane/16. elems 0..15 -> K = k0 + half*16 + e.
__device__ __forceinline__ v16bf load_a_lds(const bf16_t* buf, int row0, int lda,
                                            int k0, int lane) {
  int m = lane & 15;
  int h = (lane >> 4) << 4;
  const bf16_t* p = buf + (row0 + m) * lda + k0 + h;
  V16U u;
  u.h[0] = *(const v8bf*)p;          // 16B -> ds_load_b128
  u.h[1] = *(const v8bf*)(p + 8);    // 16B -> ds_load_b128
  return u.v;
}

// B fragment (32x16 bf16) from pre-fragmented global weight buffer:
// lane reads 16 contiguous bf16 (32 bytes).
__device__ __forceinline__ v16bf load_b_glb(const bf16_t* base, int tile, int lane) {
  return *(const v16bf*)(base + ((size_t)tile * 32 + lane) * 16);
}

// CDNA5 has a hardware TANH transcendental (ISA TRANS op set). Avoid the
// branchy OCML tanhf expansion. TRANS hazard rule: one independent op before
// the result is consumed -> keep a v_nop inside the asm fallback.
__device__ __forceinline__ float tanh_fast(float x) {
#if __has_builtin(__builtin_amdgcn_tanhf)
  return __builtin_amdgcn_tanhf(x);
#else
  float r;
  asm volatile("v_tanh_f32 %0, %1\n\tv_nop" : "=v"(r) : "v"(x));
  return r;
#endif
}

__device__ __forceinline__ float gelu_tanh(float x) {
  float x3 = x * x * x;
  return 0.5f * x * (1.0f + tanh_fast(0.7978845608028654f * (x + 0.044715f * x3)));
}
// v_exp_f32 + v_rcp_f32 instead of a precise-division chain.
__device__ __forceinline__ float sigmoidf_(float x) {
  return __builtin_amdgcn_rcpf(1.0f + __expf(-x));
}

// ---------------------------------------------------------------------------
// Weight prep: f32 [Kd, Nd] row-major -> bf16 fragment order.
// dst index i = ((kt*NT + nt)*32 + lane)*16 + e maps to
// src[(kt*32 + (lane/16)*16 + e) * Nd + nt*16 + lane%16]
// ---------------------------------------------------------------------------
__global__ void k_prep(const float* __restrict__ src, bf16_t* __restrict__ dst,
                       int Kd, int Nd) {
  int total = Kd * Nd;
  int NT = Nd >> 4;
  for (int i = blockIdx.x * blockDim.x + threadIdx.x; i < total;
       i += gridDim.x * blockDim.x) {
    int e    = i & 15;
    int L    = (i >> 4) & 31;
    int tile = i >> 9;
    int nt = tile % NT, kt = tile / NT;
    int kk = kt * 32 + ((L >> 4) << 4) + e;
    int nn = (nt << 4) + (L & 15);
    dst[i] = (bf16_t)src[(size_t)kk * Nd + nn];
  }
}

// ---------------------------------------------------------------------------
// Stage 1: per node: feat[32,320] -> gelu(feat@W1)[32,512] -> @W2 [32,128]
//          -> masked sum over 32 rows -> msum[N,128]
// ---------------------------------------------------------------------------
__global__ __launch_bounds__(256) void k_stage1(
    const float* __restrict__ localf, const float* __restrict__ pairf,
    const int* __restrict__ nbrs, const float* __restrict__ maskf,
    const bf16_t* __restrict__ W1f, const bf16_t* __restrict__ W2f,
    float* __restrict__ msum) {
  __shared__ __align__(16) bf16_t featb[KNBR * CINF];  // 20KB (aliased by h2f later)
  __shared__ __align__(16) bf16_t hb[KNBR * H1DIM];    // 32KB
  __shared__ float pmask_s[KNBR];
  float* h2f = (float*)featb;  // 32*128*4 = 16KB <= 20KB

  const int n = blockIdx.x;
  const int t = threadIdx.x;
  const int lane = t & 31, wave = t >> 5;

  if (t < KNBR) {
    int nb = nbrs[n * KNBR + t];
    pmask_s[t] = (nb != -1) ? maskf[n] : 0.0f;
  }
  for (int i = t; i < KNBR * CINF; i += 256) {
    int r = i / CINF, c = i - r * CINF;
    float v;
    if (c < LDIM)            v = localf[(size_t)n * LDIM + c];
    else if (c < 2 * LDIM) { int nb = nbrs[n * KNBR + r];
                             v = localf[(size_t)nb * LDIM + (c - LDIM)]; }
    else                     v = pairf[((size_t)n * KNBR + r) * PDIM + (c - 2 * LDIM)];
    featb[i] = (bf16_t)v;
  }
  __syncthreads();

  // GEMM1 + gelu -> hb. Output tiles: 2 (m) x 32 (n)
  for (int tile = wave; tile < 2 * (H1DIM / 16); tile += 8) {
    int mt = tile & 1, nt = tile >> 1;
    v8f acc = {};
#pragma unroll
    for (int kt = 0; kt < CINF / 32; ++kt) {
      v16bf a = load_a_lds(featb, mt * 16, CINF, kt * 32, lane);
      v16bf b = load_b_glb(W1f, kt * (H1DIM / 16) + nt, lane);
      acc = wmma_bf16(a, b, acc);
    }
    int nn = lane & 15, mb = (lane >> 4) << 3;
#pragma unroll
    for (int v = 0; v < 8; ++v)
      hb[(mt * 16 + mb + v) * H1DIM + nt * 16 + nn] = (bf16_t)gelu_tanh(acc[v]);
  }
  __syncthreads();

  // GEMM2 -> h2f. Output tiles: 2 (m) x 8 (n)
  for (int tile = wave; tile < 2 * (LDIM / 16); tile += 8) {
    int mt = tile & 1, nt = tile >> 1;
    v8f acc = {};
#pragma unroll
    for (int kt = 0; kt < H1DIM / 32; ++kt) {
      v16bf a = load_a_lds(hb, mt * 16, H1DIM, kt * 32, lane);
      v16bf b = load_b_glb(W2f, kt * (LDIM / 16) + nt, lane);
      acc = wmma_bf16(a, b, acc);
    }
    int nn = lane & 15, mb = (lane >> 4) << 3;
#pragma unroll
    for (int v = 0; v < 8; ++v)
      h2f[(mt * 16 + mb + v) * LDIM + nt * 16 + nn] = acc[v];
  }
  __syncthreads();

  if (t < LDIM) {
    float s = 0.f;
#pragma unroll
    for (int r = 0; r < KNBR; ++r) s += pmask_s[r] * h2f[r * LDIM + t];
    msum[(size_t)n * LDIM + t] = s;
  }
}

// ---------------------------------------------------------------------------
// Stage 2: 16 nodes / block. gate, LN1, gated MLP (silu), LN2.
// Writes local f32 -> d_out and bf16 copy -> ws.
// ---------------------------------------------------------------------------
__global__ __launch_bounds__(256) void k_stage2(
    const float* __restrict__ localf, const float* __restrict__ msum,
    const bf16_t* __restrict__ Wg1f, const float* __restrict__ bg1,
    const float* __restrict__ ln1s, const float* __restrict__ ln1b,
    const bf16_t* __restrict__ Wuf, const float* __restrict__ bu,
    const bf16_t* __restrict__ Wvf, const float* __restrict__ bv,
    const bf16_t* __restrict__ Wof, const float* __restrict__ bo,
    const float* __restrict__ ln2s, const float* __restrict__ ln2b,
    float* __restrict__ local_out, bf16_t* __restrict__ localb) {
  __shared__ __align__(16) float  xf[16 * LDIM];   // local -> s -> x1
  __shared__ __align__(16) bf16_t xb[16 * LDIM];
  __shared__ __align__(16) bf16_t gb[16 * H2DIM];
  __shared__ __align__(16) float  s2[16 * LDIM];

  const int n0 = blockIdx.x * 16;
  const int t = threadIdx.x, lane = t & 31, wave = t >> 5;

  for (int i = t; i < 16 * LDIM; i += 256) {
    float v = localf[(size_t)n0 * LDIM + i];
    xf[i] = v;
    xb[i] = (bf16_t)v;
  }
  __syncthreads();

  // gate GEMM [16,128]@Wg1[128,128]; wave w owns n-tile w
  {
    int nt = wave;
    v8f acc = {};
#pragma unroll
    for (int kt = 0; kt < LDIM / 32; ++kt) {
      v16bf a = load_a_lds(xb, 0, LDIM, kt * 32, lane);
      v16bf b = load_b_glb(Wg1f, kt * (LDIM / 16) + nt, lane);
      acc = wmma_bf16(a, b, acc);
    }
    int nn = lane & 15, mb = (lane >> 4) << 3;
    int col = nt * 16 + nn;
    float bgv = bg1[col];
#pragma unroll
    for (int v = 0; v < 8; ++v) {
      int m = mb + v;
      float updv = msum[(size_t)(n0 + m) * LDIM + col] * (1.0f / KNBR) *
                   sigmoidf_(acc[v] + bgv);
      xf[m * LDIM + col] += updv;  // s = local + upd
    }
  }
  __syncthreads();

  // LN1 over 128 cols: row = t/16, 8 cols per thread, shfl-reduce in 16-lane group
  {
    int row = t >> 4, sub = t & 15;
    float vals[8], s = 0.f, sq = 0.f;
#pragma unroll
    for (int j = 0; j < 8; ++j) {
      float v = xf[row * LDIM + sub * 8 + j];
      vals[j] = v; s += v; sq += v * v;
    }
#pragma unroll
    for (int m = 1; m < 16; m <<= 1) {
      s += __shfl_xor(s, m, 32);
      sq += __shfl_xor(sq, m, 32);
    }
    float mu = s * (1.0f / LDIM);
    float rstd = rsqrtf(sq * (1.0f / LDIM) - mu * mu + 1e-5f);
#pragma unroll
    for (int j = 0; j < 8; ++j) {
      int c = sub * 8 + j;
      float xv = ln1s[c] * (vals[j] - mu) * rstd + ln1b[c];
      xf[row * LDIM + c] = xv;
      xb[row * LDIM + c] = (bf16_t)xv;
    }
  }
  __syncthreads();

  // u,v GEMMs + silu gate -> gb (bf16). Wave w owns nt = w, w+8, w+16, w+24.
  for (int nt = wave; nt < H2DIM / 16; nt += 8) {
    v8f au = {}, av = {};
#pragma unroll
    for (int kt = 0; kt < LDIM / 32; ++kt) {
      v16bf a  = load_a_lds(xb, 0, LDIM, kt * 32, lane);
      v16bf bU = load_b_glb(Wuf, kt * (H2DIM / 16) + nt, lane);
      v16bf bV = load_b_glb(Wvf, kt * (H2DIM / 16) + nt, lane);
      au = wmma_bf16(a, bU, au);
      av = wmma_bf16(a, bV, av);
    }
    int nn = lane & 15, mb = (lane >> 4) << 3;
    int col = nt * 16 + nn;
    float buv = bu[col], bvv = bv[col];
#pragma unroll
    for (int v = 0; v < 8; ++v) {
      float vv = av[v] + bvv;
      float g = (vv * sigmoidf_(vv)) * (au[v] + buv);
      gb[(mb + v) * H2DIM + col] = (bf16_t)g;
    }
  }
  __syncthreads();

  // o GEMM [16,512]@Wo[512,128]; wave w owns n-tile w
  {
    int nt = wave;
    v8f acc = {};
#pragma unroll
    for (int kt = 0; kt < H2DIM / 32; ++kt) {
      v16bf a = load_a_lds(gb, 0, H2DIM, kt * 32, lane);
      v16bf b = load_b_glb(Wof, kt * (LDIM / 16) + nt, lane);
      acc = wmma_bf16(a, b, acc);
    }
    int nn = lane & 15, mb = (lane >> 4) << 3;
    int col = nt * 16 + nn;
    float bov = bo[col];
#pragma unroll
    for (int v = 0; v < 8; ++v) {
      int m = mb + v;
      s2[m * LDIM + col] = xf[m * LDIM + col] + acc[v] + bov;
    }
  }
  __syncthreads();

  // LN2 + writeback
  {
    int row = t >> 4, sub = t & 15;
    float vals[8], s = 0.f, sq = 0.f;
#pragma unroll
    for (int j = 0; j < 8; ++j) {
      float v = s2[row * LDIM + sub * 8 + j];
      vals[j] = v; s += v; sq += v * v;
    }
#pragma unroll
    for (int m = 1; m < 16; m <<= 1) {
      s += __shfl_xor(s, m, 32);
      sq += __shfl_xor(sq, m, 32);
    }
    float mu = s * (1.0f / LDIM);
    float rstd = rsqrtf(sq * (1.0f / LDIM) - mu * mu + 1e-5f);
#pragma unroll
    for (int j = 0; j < 8; ++j) {
      int c = sub * 8 + j;
      float xv = ln2s[c] * (vals[j] - mu) * rstd + ln2b[c];
      size_t gi = (size_t)(n0 + row) * LDIM + c;
      local_out[gi] = xv;
      localb[gi] = (bf16_t)xv;
    }
  }
}

// ---------------------------------------------------------------------------
// Stage 3: per node: feat2[32,320] -> gelu(@W1p)[32,128] -> @W2p [32,64]
//          * sigmoid(pair@Wg2 + bg2), residual + LN -> pair_out
// ---------------------------------------------------------------------------
__global__ __launch_bounds__(256) void k_stage3(
    const float* __restrict__ pairf, const int* __restrict__ nbrs,
    const bf16_t* __restrict__ localb,
    const bf16_t* __restrict__ W1pf, const bf16_t* __restrict__ W2pf,
    const bf16_t* __restrict__ Wg2f, const float* __restrict__ bg2,
    const float* __restrict__ ln3s, const float* __restrict__ ln3b,
    float* __restrict__ pair_out) {
  __shared__ __align__(16) bf16_t featb[KNBR * CINF];  // 20KB
  __shared__ __align__(16) bf16_t pairb[KNBR * PDIM];  // 4KB
  __shared__ __align__(16) bf16_t hb[KNBR * HPDIM];    // 8KB
  __shared__ __align__(16) float  hpf[KNBR * PDIM];    // 8KB
  __shared__ __align__(16) float  glf[KNBR * PDIM];    // 8KB

  const int n = blockIdx.x;
  const int t = threadIdx.x, lane = t & 31, wave = t >> 5;

  for (int i = t; i < KNBR * CINF; i += 256) {
    int r = i / CINF, c = i - r * CINF;
    bf16_t v;
    if (c < LDIM)            v = localb[(size_t)n * LDIM + c];
    else if (c < 2 * LDIM) { int nb = nbrs[n * KNBR + r];
                             v = localb[(size_t)nb * LDIM + (c - LDIM)]; }
    else {
      float pv = pairf[((size_t)n * KNBR + r) * PDIM + (c - 2 * LDIM)];
      v = (bf16_t)pv;
      pairb[r * PDIM + (c - 2 * LDIM)] = v;
    }
    featb[i] = v;
  }
  __syncthreads();

  // GEMM1 + gelu -> hb. Tiles 2 x 8; wave w owns w and w+8.
  for (int tile = wave; tile < 2 * (HPDIM / 16); tile += 8) {
    int mt = tile & 1, nt = tile >> 1;
    v8f acc = {};
#pragma unroll
    for (int kt = 0; kt < CINF / 32; ++kt) {
      v16bf a = load_a_lds(featb, mt * 16, CINF, kt * 32, lane);
      v16bf b = load_b_glb(W1pf, kt * (HPDIM / 16) + nt, lane);
      acc = wmma_bf16(a, b, acc);
    }
    int nn = lane & 15, mb = (lane >> 4) << 3;
#pragma unroll
    for (int v = 0; v < 8; ++v)
      hb[(mt * 16 + mb + v) * HPDIM + nt * 16 + nn] = (bf16_t)gelu_tanh(acc[v]);
  }
  // gate GEMM pair[32,64]@Wg2[64,64] -> glf. 8 tiles, one per wave.
  {
    int mt = wave & 1, nt = wave >> 1;
    v8f acc = {};
#pragma unroll
    for (int kt = 0; kt < PDIM / 32; ++kt) {
      v16bf a = load_a_lds(pairb, mt * 16, PDIM, kt * 32, lane);
      v16bf b = load_b_glb(Wg2f, kt * (PDIM / 16) + nt, lane);
      acc = wmma_bf16(a, b, acc);
    }
    int nn = lane & 15, mb = (lane >> 4) << 3;
#pragma unroll
    for (int v = 0; v < 8; ++v)
      glf[(mt * 16 + mb + v) * PDIM + nt * 16 + nn] = acc[v];
  }
  __syncthreads();

  // GEMM2 [32,128]@W2p[128,64] -> hpf. 8 tiles, one per wave.
  {
    int mt = wave & 1, nt = wave >> 1;
    v8f acc = {};
#pragma unroll
    for (int kt = 0; kt < HPDIM / 32; ++kt) {
      v16bf a = load_a_lds(hb, mt * 16, HPDIM, kt * 32, lane);
      v16bf b = load_b_glb(W2pf, kt * (PDIM / 16) + nt, lane);
      acc = wmma_bf16(a, b, acc);
    }
    int nn = lane & 15, mb = (lane >> 4) << 3;
#pragma unroll
    for (int v = 0; v < 8; ++v)
      hpf[(mt * 16 + mb + v) * PDIM + nt * 16 + nn] = acc[v];
  }
  __syncthreads();

  // gate + residual + LN over 64 cols. row = t/8, 8 cols per thread.
  {
    int row = t >> 3, sub = t & 7;
    float vals[8], s = 0.f, sq = 0.f;
#pragma unroll
    for (int j = 0; j < 8; ++j) {
      int c = sub * 8 + j;
      float hp = hpf[row * PDIM + c] * sigmoidf_(glf[row * PDIM + c] + bg2[c]);
      float v = pairf[((size_t)n * KNBR + row) * PDIM + c] + hp;
      vals[j] = v; s += v; sq += v * v;
    }
#pragma unroll
    for (int m = 1; m < 8; m <<= 1) {
      s += __shfl_xor(s, m, 32);
      sq += __shfl_xor(sq, m, 32);
    }
    float mu = s * (1.0f / PDIM);
    float rstd = rsqrtf(sq * (1.0f / PDIM) - mu * mu + 1e-5f);
#pragma unroll
    for (int j = 0; j < 8; ++j) {
      int c = sub * 8 + j;
      pair_out[((size_t)n * KNBR + row) * PDIM + c] =
          ln3s[c] * (vals[j] - mu) * rstd + ln3b[c];
    }
  }
}

// ---------------------------------------------------------------------------
extern "C" void kernel_launch(void* const* d_in, const int* in_sizes, int n_in,
                              void* d_out, int out_size, void* d_ws, size_t ws_size,
                              hipStream_t stream) {
  const float* localf = (const float*)d_in[0];
  const float* pairf  = (const float*)d_in[1];
  const int*   nbrs   = (const int*)d_in[2];
  const float* maskf  = (const float*)d_in[3];
  const float* W1   = (const float*)d_in[4];
  const float* W2   = (const float*)d_in[5];
  const float* Wg1  = (const float*)d_in[6];
  const float* bg1  = (const float*)d_in[7];
  const float* ln1s = (const float*)d_in[8];
  const float* ln1b = (const float*)d_in[9];
  const float* Wu   = (const float*)d_in[10];
  const float* bu   = (const float*)d_in[11];
  const float* Wv   = (const float*)d_in[12];
  const float* bv   = (const float*)d_in[13];
  const float* Wo   = (const float*)d_in[14];
  const float* bo   = (const float*)d_in[15];
  const float* ln2s = (const float*)d_in[16];
  const float* ln2b = (const float*)d_in[17];
  const float* W1p  = (const float*)d_in[18];
  const float* W2p  = (const float*)d_in[19];
  const float* Wg2  = (const float*)d_in[20];
  const float* bg2  = (const float*)d_in[21];
  const float* ln3s = (const float*)d_in[22];
  const float* ln3b = (const float*)d_in[23];

  char* ws = (char*)d_ws;
  size_t off = 0;
  float*  msum   = (float*)(ws + off);  off += (size_t)NNODES * LDIM * 4;
  bf16_t* localb = (bf16_t*)(ws + off); off += (size_t)NNODES * LDIM * 2;
  bf16_t* W1f  = (bf16_t*)(ws + off); off += (size_t)CINF  * H1DIM * 2;
  bf16_t* W2f  = (bf16_t*)(ws + off); off += (size_t)H1DIM * LDIM  * 2;
  bf16_t* Wg1f = (bf16_t*)(ws + off); off += (size_t)LDIM  * LDIM  * 2;
  bf16_t* Wuf  = (bf16_t*)(ws + off); off += (size_t)LDIM  * H2DIM * 2;
  bf16_t* Wvf  = (bf16_t*)(ws + off); off += (size_t)LDIM  * H2DIM * 2;
  bf16_t* Wof  = (bf16_t*)(ws + off); off += (size_t)H2DIM * LDIM  * 2;
  bf16_t* W1pf = (bf16_t*)(ws + off); off += (size_t)CINF  * HPDIM * 2;
  bf16_t* W2pf = (bf16_t*)(ws + off); off += (size_t)HPDIM * PDIM  * 2;
  bf16_t* Wg2f = (bf16_t*)(ws + off); off += (size_t)PDIM  * PDIM  * 2;
  (void)off; (void)ws_size; (void)in_sizes; (void)n_in; (void)out_size;

  auto prep = [&](const float* s, bf16_t* d, int Kd, int Nd) {
    int total = Kd * Nd;
    k_prep<<<(total + 255) / 256, 256, 0, stream>>>(s, d, Kd, Nd);
  };
  prep(W1,  W1f,  CINF,  H1DIM);
  prep(W2,  W2f,  H1DIM, LDIM);
  prep(Wg1, Wg1f, LDIM,  LDIM);
  prep(Wu,  Wuf,  LDIM,  H2DIM);
  prep(Wv,  Wvf,  LDIM,  H2DIM);
  prep(Wo,  Wof,  H2DIM, LDIM);
  prep(W1p, W1pf, CINF,  HPDIM);
  prep(W2p, W2pf, HPDIM, PDIM);
  prep(Wg2, Wg2f, PDIM,  PDIM);

  k_stage1<<<NNODES, 256, 0, stream>>>(localf, pairf, nbrs, maskf, W1f, W2f, msum);

  k_stage2<<<NNODES / 16, 256, 0, stream>>>(
      localf, msum, Wg1f, bg1, ln1s, ln1b, Wuf, bu, Wvf, bv, Wof, bo, ln2s, ln2b,
      (float*)d_out, localb);

  k_stage3<<<NNODES, 256, 0, stream>>>(
      pairf, nbrs, localb, W1pf, W2pf, Wg2f, bg2, ln3s, ln3b,
      (float*)d_out + (size_t)NNODES * LDIM);
}
